// AddAttention_85676007620702
// MI455X (gfx1250) — matrix operand order repaired and checked by reading, not compiled, not run on hardware
//
#include <hip/hip_runtime.h>

typedef __bf16 bf16_t;
typedef __attribute__((ext_vector_type(16))) __bf16 v16bf;
typedef __attribute__((ext_vector_type(8)))  __bf16 v8bf;
typedef __attribute__((ext_vector_type(8)))  float   v8f;

#define B_ 32
#define L_ 512
#define D_ 1024

#define LDS_STRIDE 40   // bf16 elements per LDS row (20 banks -> conflict-free frag reads)

// ---- CDNA5 async global->LDS (ASYNCcnt-tracked DMA, GV addressing mode) ----
// VDST = VGPR with LDS byte offset; VADDR = 64-bit global address.
__device__ __forceinline__ void async_ld_b128(unsigned lds_off, const void* gaddr) {
  asm volatile("global_load_async_to_lds_b128 %0, %1, off"
               :: "v"(lds_off), "v"((unsigned long long)(uintptr_t)gaddr)
               : "memory");
}
__device__ __forceinline__ void wait_async0() {
  asm volatile("s_wait_asynccnt 0x0" ::: "memory");
}

// ---------------- fragment loaders (CDNA5 wave32 16-bit layouts) ----------------
// A-matrix 16x32 (MxK). Lane l (0-15): row m=l, K [0..7],[16..23]; lane l+16: K [8..15],[24..31].
__device__ __forceinline__ v16bf load_a_frag_lds(const bf16_t* A /*row0 of tile*/, int lane) {
  const int l  = lane & 15;
  const int hi = lane >> 4;
  const bf16_t* p = A + l * LDS_STRIDE + hi * 8;
  v8bf lo = *(const v8bf*)(p);
  v8bf hv = *(const v8bf*)(p + 16);
  v16bf r;
  #pragma unroll
  for (int i = 0; i < 8; ++i) { r[i] = lo[i]; r[i + 8] = hv[i]; }
  return r;
}
// B-matrix 32x16 (KxN) stored as B^T rows. Lane n (0-15): col n, K [0..7],[16..23]; lane n+16: K [8..15],[24..31].
__device__ __forceinline__ v16bf load_b_frag_lds(const bf16_t* Bt /*row0 of tile*/, int lane) {
  const int n  = lane & 15;
  const int hi = lane >> 4;
  const bf16_t* p = Bt + n * LDS_STRIDE + hi * 8;
  v8bf lo = *(const v8bf*)(p);
  v8bf hv = *(const v8bf*)(p + 16);
  v16bf r;
  #pragma unroll
  for (int i = 0; i < 8; ++i) { r[i] = lo[i]; r[i + 8] = hv[i]; }
  return r;
}

// ---------------- generic WMMA GEMM with double-buffered async LDS staging ----------------
// C[m][n] = sum_k A[m][k] * Bt[n][k]   (Bt is [N][K]; all reference weights already are)
// A may be split along K at kSplit between two sources (concat-K GEMMs).
// EPI: 0 = store f32 ; 1 = tanh(acc+bias0[n]+bias1[n]) -> bf16 ; 2 = store bf16 ; 3 = acc+bias0[n] -> f32
template<int EPI>
__global__ __launch_bounds__(256) void wmma_gemm(
    const bf16_t* __restrict__ A0, int lda0, long aStride0,
    const bf16_t* __restrict__ A1, int lda1, long aStride1, int kSplit,
    const bf16_t* __restrict__ Bw, int ldb, long bStride,
    int M, int N, int K,
    const float* __restrict__ bias0, const float* __restrict__ bias1,
    float* __restrict__ outF, bf16_t* __restrict__ outB, int ldo, long oStride)
{
  __shared__ bf16_t sA[2][64][LDS_STRIDE];    //  64 rows of A tile (k-step 32)
  __shared__ bf16_t sB[2][256][LDS_STRIDE];   // 256 rows of B^T tile

  const int tid  = threadIdx.x;
  const int lane = tid & 31;
  const int wave = tid >> 5;
  const int wm = wave >> 2;      // 0..1 -> 32 rows per wave
  const int wn = wave & 3;       // 0..3 -> 64 cols per wave
  const long bz = blockIdx.z;
  const int m0 = blockIdx.y * 64;
  const int n0 = blockIdx.x * 256;

  const bf16_t* A0b = A0 + bz * aStride0;
  const bf16_t* A1b = A1 ? (A1 + bz * aStride1) : (const bf16_t*)0;
  const bf16_t* Bb  = Bw + bz * bStride;

  // --- stager mapping: each thread DMAs 4x16B of B (one full row) + 1x16B of A ---
  const int arow = tid >> 2;        // 0..63
  const int achk = tid & 3;         // 0..3  (16B chunk within 64B k-row)
  const char* gB  = (const char*)(Bb + (size_t)(n0 + tid) * ldb);            // + 2*k
  const char* gA0 = (const char*)(A0b + (size_t)(m0 + arow) * lda0) + achk * 16;
  const char* gA1 = A1b ? (const char*)(A1b + (size_t)(m0 + arow) * lda1) + achk * 16
                        : (const char*)0;
  unsigned ldsB[2] = { (unsigned)(uintptr_t)&sB[0][tid][0],
                       (unsigned)(uintptr_t)&sB[1][tid][0] };
  unsigned ldsA[2] = { (unsigned)(uintptr_t)&sA[0][arow][achk * 8],
                       (unsigned)(uintptr_t)&sA[1][arow][achk * 8] };

  auto issue = [&](int k, int buf) {
    #pragma unroll
    for (int c = 0; c < 4; ++c)
      async_ld_b128(ldsB[buf] + c * 16, gB + 2 * k + c * 16);
    if (k < kSplit) async_ld_b128(ldsA[buf], gA0 + 2 * k);
    else            async_ld_b128(ldsA[buf], gA1 + 2 * (k - kSplit));
  };

  v8f acc[2][4] = {};

  issue(0, 0);
  for (int k = 0; k < K; k += 32) {
    const int buf = (k >> 5) & 1;
    wait_async0();            // my DMA for tile `buf` landed in LDS
    __syncthreads();          // everyone's tile landed; prev tile fully consumed
    if (k + 32 < K) issue(k + 32, buf ^ 1);

    v16bf a[2], b[4];
    #pragma unroll
    for (int i = 0; i < 2; ++i)
      a[i] = load_a_frag_lds(&sA[buf][wm * 32 + i * 16][0], lane);
    #pragma unroll
    for (int j = 0; j < 4; ++j)
      b[j] = load_b_frag_lds(&sB[buf][wn * 64 + j * 16][0], lane);

    #pragma unroll
    for (int i = 0; i < 2; ++i)
      #pragma unroll
      for (int j = 0; j < 4; ++j)
        acc[i][j] = __builtin_amdgcn_wmma_f32_16x16x32_bf16(
            false, a[i], false, b[j], (short)0, acc[i][j], false, false);
  }

  // C/D layout: VGPR r -> lanes 0-15: (M=r, N=lane); lanes 16-31: (M=r+8, N=lane-16)
  const int l  = lane & 15;
  const int hi = lane >> 4;
  #pragma unroll
  for (int i = 0; i < 2; ++i)
    #pragma unroll
    for (int j = 0; j < 4; ++j)
      #pragma unroll
      for (int r = 0; r < 8; ++r) {
        const int m = m0 + wm * 32 + i * 16 + r + 8 * hi;
        const int n = n0 + wn * 64 + j * 16 + l;
        float v = acc[i][j][r];
        if (EPI == 1) v = tanhf(v + bias0[n] + bias1[n]);
        if (EPI == 3) v += bias0[n];
        const long off = bz * oStride + (long)m * ldo + n;
        if (EPI == 0 || EPI == 3) outF[off] = v;
        else                      outB[off] = (bf16_t)v;
      }
}

// ---------------- prep kernels ----------------
__global__ void pack2_bf16(const float* __restrict__ X, const float* __restrict__ Y,
                           bf16_t* __restrict__ dst, int cols, long total) {
  long i = (long)blockIdx.x * blockDim.x + threadIdx.x;
  if (i >= total) return;
  const int w2 = cols * 2;
  const long m = i / w2;
  const int  k = (int)(i - m * w2);
  const float v = (k < cols) ? X[m * (long)cols + k] : Y[m * (long)cols + (k - cols)];
  dst[i] = (bf16_t)v;
}

__global__ void cvt_bf16(const float* __restrict__ X, bf16_t* __restrict__ dst, long n) {
  long i = (long)blockIdx.x * blockDim.x + threadIdx.x;
  if (i < n) dst[i] = (bf16_t)X[i];
}

// ctxT[b][d][w] = (bf16) context[b][w][d]
__global__ __launch_bounds__(256) void transpose_ctx_bf16(const float* __restrict__ ctx,
                                                          bf16_t* __restrict__ ctxT) {
  __shared__ bf16_t tile[32][33];
  const int b  = blockIdx.z;
  const int w0 = blockIdx.x * 32;
  const int d0 = blockIdx.y * 32;
  const int tx = threadIdx.x;
  const int ty = threadIdx.y;
  #pragma unroll
  for (int i = 0; i < 32; i += 8)
    tile[ty + i][tx] = (bf16_t)ctx[((long)b * L_ + (w0 + ty + i)) * D_ + d0 + tx];
  __syncthreads();
  #pragma unroll
  for (int i = 0; i < 32; i += 8)
    ctxT[((long)b * D_ + (d0 + ty + i)) * L_ + w0 + tx] = tile[tx][ty + i];
}

// row softmax over 512 logits; fp32 alpha (returned) + bf16 alpha (for mix GEMM)
__global__ __launch_bounds__(256) void softmax_rows(const float* __restrict__ logits,
                                                    float* __restrict__ alphaF,
                                                    bf16_t* __restrict__ alphaB) {
  const long row = blockIdx.x;
  const float* p = logits + row * (long)L_;
  const int t = threadIdx.x;
  float x0 = p[t], x1 = p[t + 256];
  __shared__ float red[256];
  red[t] = fmaxf(x0, x1);
  __syncthreads();
  for (int s = 128; s > 0; s >>= 1) {
    if (t < s) red[t] = fmaxf(red[t], red[t + s]);
    __syncthreads();
  }
  const float rowmax = red[0];
  __syncthreads();
  const float e0 = __expf(x0 - rowmax), e1 = __expf(x1 - rowmax);
  red[t] = e0 + e1;
  __syncthreads();
  for (int s = 128; s > 0; s >>= 1) {
    if (t < s) red[t] += red[t + s];
    __syncthreads();
  }
  const float inv = 1.0f / red[0];
  const float a0 = e0 * inv, a1 = e1 * inv;
  alphaF[row * L_ + t]       = a0;
  alphaF[row * L_ + t + 256] = a1;
  alphaB[row * L_ + t]       = (bf16_t)a0;
  alphaB[row * L_ + t + 256] = (bf16_t)a1;
}

// ---------------- host launch ----------------
extern "C" void kernel_launch(void* const* d_in, const int* in_sizes, int n_in,
                              void* d_out, int out_size, void* d_ws, size_t ws_size,
                              hipStream_t stream) {
  const float* query   = (const float*)d_in[0];
  const float* context = (const float*)d_in[1];
  const float* W_in    = (const float*)d_in[2];
  const float* b_in    = (const float*)d_in[3];
  const float* W_in2   = (const float*)d_in[4];
  const float* b_in2   = (const float*)d_in[5];
  const float* W_out   = (const float*)d_in[6];
  const float* W_fc    = (const float*)d_in[7];
  const float* b_fc    = (const float*)d_in[8];

  float* out   = (float*)d_out;                 // [B,L,D]
  float* alpha = out + (long)B_ * L_ * D_;      // [B,L,L]

  char* w = (char*)d_ws;
  bf16_t* Abuf  = (bf16_t*)w; w += (size_t)(B_ * L_) * (2 * D_) * 2;  // [16384][2048] = query|context
  bf16_t* WB1   = (bf16_t*)w; w += (size_t)D_ * (2 * D_) * 2;         // [1024][2048]  = W_in|W_in2
  bf16_t* Woutb = (bf16_t*)w; w += (size_t)L_ * D_ * 2;               // [512][1024]
  bf16_t* Wfcb  = (bf16_t*)w; w += (size_t)D_ * (2 * D_) * 2;         // [1024][2048]
  bf16_t* Tbf   = (bf16_t*)w; w += (size_t)(B_ * L_) * D_ * 2;        // tanh features
  float*  logit = (float*)w;  w += (size_t)B_ * L_ * L_ * 4;          // attention logits
  bf16_t* alphB = (bf16_t*)w; w += (size_t)B_ * L_ * L_ * 2;          // alpha bf16
  bf16_t* ctxT  = (bf16_t*)w; w += (size_t)B_ * D_ * L_ * 2;          // context^T bf16
  bf16_t* mixb  = (bf16_t*)w; w += (size_t)(B_ * L_) * D_ * 2;        // mix bf16

  { long tot = (long)(B_ * L_) * (2 * D_);
    pack2_bf16<<<dim3((unsigned)((tot + 255) / 256)), 256, 0, stream>>>(query, context, Abuf, D_, tot); }
  { long tot = (long)D_ * (2 * D_);
    pack2_bf16<<<dim3((unsigned)((tot + 255) / 256)), 256, 0, stream>>>(W_in, W_in2, WB1, D_, tot); }
  { long tot = (long)L_ * D_;
    cvt_bf16<<<dim3((unsigned)((tot + 255) / 256)), 256, 0, stream>>>(W_out, Woutb, tot); }
  { long tot = (long)D_ * (2 * D_);
    cvt_bf16<<<dim3((unsigned)((tot + 255) / 256)), 256, 0, stream>>>(W_fc, Wfcb, tot); }
  transpose_ctx_bf16<<<dim3(L_ / 32, D_ / 32, B_), dim3(32, 8), 0, stream>>>(context, ctxT);

  // GEMM1: T = tanh([q|c] @ [W_in|W_in2]^T + b_in + b_in2)   M=16384 N=1024 K=2048
  wmma_gemm<1><<<dim3(D_ / 256, (B_ * L_) / 64, 1), 256, 0, stream>>>(
      Abuf, 2 * D_, 0, nullptr, 0, 0, 2 * D_,
      WB1, 2 * D_, 0,
      B_ * L_, D_, 2 * D_,
      b_in, b_in2, nullptr, Tbf, D_, 0);

  // GEMM2: logits = T @ W_out^T   per batch M=512 N=512 K=1024
  wmma_gemm<0><<<dim3(L_ / 256, L_ / 64, B_), 256, 0, stream>>>(
      Tbf, D_, (long)L_ * D_, nullptr, 0, 0, D_,
      Woutb, D_, 0,
      L_, L_, D_,
      nullptr, nullptr, logit, nullptr, L_, (long)L_ * L_);

  softmax_rows<<<dim3(B_ * L_), 256, 0, stream>>>(logit, alpha, alphB);

  // GEMM3: mix = alpha @ context   per batch M=512 N=1024 K=512 (B operand = ctxT)
  wmma_gemm<2><<<dim3(D_ / 256, L_ / 64, B_), 256, 0, stream>>>(
      alphB, L_, (long)L_ * L_, nullptr, 0, 0, L_,
      ctxT, L_, (long)D_ * L_,
      L_, D_, L_,
      nullptr, nullptr, nullptr, mixb, D_, (long)L_ * D_);

  // GEMM4: out = [mix|context] @ W_fc^T + b_fc   per batch M=512 N=1024 K=2048
  wmma_gemm<3><<<dim3(D_ / 256, L_ / 64, B_), 256, 0, stream>>>(
      mixb, D_, (long)L_ * D_, Abuf + D_, 2 * D_, (long)L_ * 2 * D_, D_,
      Wfcb, 2 * D_, 0,
      L_, D_, 2 * D_,
      b_fc, nullptr, out, nullptr, D_, (long)L_ * D_);
}
